// TimeSeriesRandomForest_3951369912735
// MI455X (gfx1250) — matrix operand order repaired and checked by reading, not compiled, not run on hardware
//
#include <hip/hip_runtime.h>
#include <hip/hip_bf16.h>

// ---------------------------------------------------------------------------
// Fused 100-estimator MLP forest, bf16 WMMA path for MI455X (gfx1250, wave32)
// Round 2: hoist loop-invariant A-matrix fragments into registers so each
// wave reads its x / h1 rows from LDS once instead of once per N-tile.
// ---------------------------------------------------------------------------

typedef __attribute__((ext_vector_type(16))) __bf16        v16bf;
typedef __attribute__((ext_vector_type(8)))  float         v8f;
typedef __attribute__((ext_vector_type(4)))  unsigned int  v4u;

#define BATCH   16384
#define NFEAT   128
#define NEST    100
#define H1D     64
#define H2D     32
#define TB      64          // batch rows per block
#define NTHR    128         // 4 waves of 32

// padded LDS row strides (in bf16 elements); all row offsets stay 16B aligned
#define XT_S    136
#define W1_S    136
#define H1_S    72
#define W2_S    72
#define H2_S    40

__device__ __forceinline__ unsigned short f2bf(float f) {
  unsigned u = __float_as_uint(f);
  u += 0x7FFFu + ((u >> 16) & 1u);          // round-to-nearest-even
  return (unsigned short)(u >> 16);
}
__device__ __forceinline__ float bf2f(unsigned short h) {
  return __uint_as_float(((unsigned)h) << 16);
}

struct FragBits { v4u lo, hi; };
__device__ __forceinline__ v16bf make_frag(const unsigned short* p0,
                                           const unsigned short* p1) {
  FragBits f;
  f.lo = *(const v4u*)p0;
  f.hi = *(const v4u*)p1;
  return __builtin_bit_cast(v16bf, f);
}

// -------------------------- fp32 -> bf16 staging ---------------------------
__global__ void cvt_bf16(const float* __restrict__ src,
                         unsigned short* __restrict__ dst, int n) {
  int i = blockIdx.x * blockDim.x + threadIdx.x;
  if (i < n) dst[i] = f2bf(src[i]);
}

// ------------------------------ fused forest -------------------------------
__global__ __launch_bounds__(NTHR)
void forest_fused(const unsigned short* __restrict__ xbf,
                  const unsigned short* __restrict__ w1bf,
                  const float* __restrict__ b1,
                  const unsigned short* __restrict__ w2bf,
                  const float* __restrict__ b2,
                  const float* __restrict__ w3,
                  const float* __restrict__ b3,
                  float* __restrict__ partial) {
  __shared__ unsigned short sX [TB  * XT_S];
  __shared__ unsigned short sW1[H1D * W1_S];
  __shared__ unsigned short sH1[TB  * H1_S];
  __shared__ unsigned short sW2[H2D * W2_S];
  __shared__ unsigned short sH2[TB  * H2_S];

  const int tid = threadIdx.x;
  const int e   = blockIdx.y;
  const int b0  = blockIdx.x * TB;

  // ---- stage x tile [TB x 128] (bf16) ----
  for (int j = tid; j < TB * (NFEAT / 8); j += NTHR) {
    int row = j >> 4, c8 = (j & 15) << 3;
    *(v4u*)&sX[row * XT_S + c8] =
        *(const v4u*)&xbf[(size_t)(b0 + row) * NFEAT + c8];
  }
  // ---- stage W1[e] [64 x 128] ----
  const unsigned short* w1e = w1bf + (size_t)e * H1D * NFEAT;
  for (int j = tid; j < H1D * (NFEAT / 8); j += NTHR) {
    int row = j >> 4, c8 = (j & 15) << 3;
    *(v4u*)&sW1[row * W1_S + c8] = *(const v4u*)&w1e[row * NFEAT + c8];
  }
  // ---- stage W2[e] [32 x 64] ----
  const unsigned short* w2e = w2bf + (size_t)e * H2D * H1D;
  for (int j = tid; j < H2D * (H1D / 8); j += NTHR) {
    int row = j >> 3, c8 = (j & 7) << 3;
    *(v4u*)&sW2[row * W2_S + c8] = *(const v4u*)&w2e[row * H1D + c8];
  }
  __syncthreads();

  const int lane  = tid & 31;
  const int wv    = tid >> 5;        // wave id == M-tile (0..3)
  const int lrow  = lane & 15;
  const int khalf = lane >> 4;       // 0 or 1
  const int arow  = wv * 16 + lrow;  // A-matrix row held by this lane

  // =================== GEMM1: h1 = relu(x @ W1^T + b1) ====================
  // Hoist A fragments: invariant across all 4 N-tiles.
  v16bf a1[NFEAT / 32];
  #pragma unroll
  for (int kt = 0; kt < NFEAT / 32; ++kt) {
    const int kb = kt * 32;
    a1[kt] = make_frag(&sX[arow * XT_S + kb + 8 * khalf],
                       &sX[arow * XT_S + kb + 8 * khalf + 16]);
  }
  #pragma unroll
  for (int nt = 0; nt < H1D / 16; ++nt) {
    v8f acc = {};
    const int ocol = nt * 16 + lrow;           // output column / W1 row
    #pragma unroll
    for (int kt = 0; kt < NFEAT / 32; ++kt) {
      const int kb = kt * 32;
      // B lane layout: col = lane%16, K = kb + 8*khalf + [0..7], kb + 16 + 8*khalf + [0..7]
      v16bf b = make_frag(&sW1[ocol * W1_S + kb + 8 * khalf],
                          &sW1[ocol * W1_S + kb + 8 * khalf + 16]);
      acc = __builtin_amdgcn_wmma_f32_16x16x32_bf16(
          false, a1[kt], false, b, (short)0, acc, false, false);
    }
    const float bias = b1[e * H1D + ocol];
    #pragma unroll
    for (int r = 0; r < 8; ++r) {              // C: VGPR r -> row r+8*khalf
      float v = acc[r] + bias;
      v = v > 0.f ? v : 0.f;
      sH1[(wv * 16 + r + 8 * khalf) * H1_S + ocol] = f2bf(v);
    }
  }
  __syncthreads();

  // =================== GEMM2: h2 = relu(h1 @ W2^T + b2) ===================
  v16bf a2[H1D / 32];
  #pragma unroll
  for (int kt = 0; kt < H1D / 32; ++kt) {
    const int kb = kt * 32;
    a2[kt] = make_frag(&sH1[arow * H1_S + kb + 8 * khalf],
                       &sH1[arow * H1_S + kb + 8 * khalf + 16]);
  }
  #pragma unroll
  for (int nt = 0; nt < H2D / 16; ++nt) {
    v8f acc = {};
    const int ocol = nt * 16 + lrow;
    #pragma unroll
    for (int kt = 0; kt < H1D / 32; ++kt) {
      const int kb = kt * 32;
      v16bf b = make_frag(&sW2[ocol * W2_S + kb + 8 * khalf],
                          &sW2[ocol * W2_S + kb + 8 * khalf + 16]);
      acc = __builtin_amdgcn_wmma_f32_16x16x32_bf16(
          false, a2[kt], false, b, (short)0, acc, false, false);
    }
    const float bias = b2[e * H2D + ocol];
    #pragma unroll
    for (int r = 0; r < 8; ++r) {
      float v = acc[r] + bias;
      v = v > 0.f ? v : 0.f;
      sH2[(wv * 16 + r + 8 * khalf) * H2_S + ocol] = f2bf(v);
    }
  }
  __syncthreads();

  // ============== layer 3: per-row dot-32, deterministic write ============
  if (tid < TB) {
    const float* w3e = w3 + e * H2D;
    float acc = 0.f;
    #pragma unroll
    for (int o = 0; o < H2D; ++o)
      acc += bf2f(sH2[tid * H2_S + o]) * w3e[o];
    partial[(size_t)e * BATCH + b0 + tid] = acc + b3[e];
  }
}

// --------------------------- mean over estimators --------------------------
__global__ void reduce_mean(const float* __restrict__ partial,
                            float* __restrict__ out) {
  int b = blockIdx.x * blockDim.x + threadIdx.x;
  if (b < BATCH) {
    float s = 0.f;
    for (int e = 0; e < NEST; ++e) s += partial[(size_t)e * BATCH + b];
    out[b] = s * (1.0f / NEST);
  }
}

// ------------------------------- launcher ----------------------------------
extern "C" void kernel_launch(void* const* d_in, const int* in_sizes, int n_in,
                              void* d_out, int out_size, void* d_ws, size_t ws_size,
                              hipStream_t stream) {
  const float* x  = (const float*)d_in[0];   // [16384,128]
  const float* W1 = (const float*)d_in[1];   // [100,64,128]
  const float* b1 = (const float*)d_in[2];   // [100,64]
  const float* W2 = (const float*)d_in[3];   // [100,32,64]
  const float* b2 = (const float*)d_in[4];   // [100,32]
  const float* W3 = (const float*)d_in[5];   // [100,1,32]
  const float* b3 = (const float*)d_in[6];   // [100,1]
  float* out = (float*)d_out;

  char* ws = (char*)d_ws;
  const size_t xbf_off  = 0;                            // 16384*128*2 = 4 MiB
  const size_t w1_off   = xbf_off + (size_t)BATCH * NFEAT * 2;
  const size_t w2_off   = w1_off  + (size_t)NEST * H1D * NFEAT * 2;
  const size_t part_off = w2_off  + (size_t)NEST * H2D * H1D * 2;
  unsigned short* xbf  = (unsigned short*)(ws + xbf_off);
  unsigned short* w1bf = (unsigned short*)(ws + w1_off);
  unsigned short* w2bf = (unsigned short*)(ws + w2_off);
  float* partial       = (float*)(ws + part_off);       // [100][16384]

  const int n_x  = BATCH * NFEAT;
  const int n_w1 = NEST * H1D * NFEAT;
  const int n_w2 = NEST * H2D * H1D;
  cvt_bf16<<<(n_x  + 255) / 256, 256, 0, stream>>>(x,  xbf,  n_x);
  cvt_bf16<<<(n_w1 + 255) / 256, 256, 0, stream>>>(W1, w1bf, n_w1);
  cvt_bf16<<<(n_w2 + 255) / 256, 256, 0, stream>>>(W2, w2bf, n_w2);

  dim3 grid(BATCH / TB, NEST);   // 256 x 100 blocks
  forest_fused<<<grid, NTHR, 0, stream>>>(xbf, w1bf, b1, w2bf, b2, W3, b3,
                                          partial);

  reduce_mean<<<(BATCH + 255) / 256, 256, 0, stream>>>(partial, out);
}
